// Model_39676907885623
// MI455X (gfx1250) — matrix operand order repaired and checked by reading, not compiled, not run on hardware
//
#include <hip/hip_runtime.h>

// Problem constants (from reference)
#define BB 4
#define CC 128
#define NHD 2
#define DD 64
#define NN 4096
#define QSCALE 0.29730177875068026f   // 128^-0.25 folded into q at projection time

typedef __attribute__((ext_vector_type(16))) __bf16 v16bf;
typedef __attribute__((ext_vector_type(8)))  float  v8f;

#ifndef __has_builtin
#define __has_builtin(x) 0
#endif

#if __has_builtin(__builtin_amdgcn_global_load_async_to_lds_b128)
#define USE_ASYNC_LDS 1
#else
#define USE_ASYNC_LDS 0
#endif

__device__ __forceinline__ unsigned short f2bfu(float f) {
  unsigned u = __float_as_uint(f);
  return (unsigned short)((u + 0x7FFFu + ((u >> 16) & 1u)) >> 16);
}
__device__ __forceinline__ __bf16 f2bf(float f) {
  unsigned short s = f2bfu(f);
  __bf16 b;
  __builtin_memcpy(&b, &s, 2);
  return b;
}

#if USE_ASYNC_LDS
// Builtin signature (from the compiler diagnostic): first arg is
// "int __vector(4) __device__ *" i.e. addrspace(1) int4*; second is the
// addrspace(3) counterpart. Cast via integers (low 32 bits of a generic LDS
// pointer are the LDS offset; global generic == flat address).
typedef int v4i_vs __attribute__((vector_size(16)));
typedef v4i_vs __attribute__((address_space(1)))* gbl_v4i;
typedef v4i_vs __attribute__((address_space(3)))* lds_v4i;

__device__ __forceinline__ void async_copy16(const __bf16* g, __bf16* l) {
  gbl_v4i gp = (gbl_v4i)(unsigned long long)(uintptr_t)g;
  lds_v4i lp = (lds_v4i)(unsigned)(uintptr_t)l;
  __builtin_amdgcn_global_load_async_to_lds_b128(gp, lp, 0, 0);
}
__device__ __forceinline__ void wait_async0() {
#if __has_builtin(__builtin_amdgcn_s_wait_asynccnt)
  __builtin_amdgcn_s_wait_asynccnt(0);
#else
  asm volatile("s_wait_asynccnt 0x0" ::: "memory");
#endif
}
#endif

// Two ds_load_b128 -> one WMMA A fragment (row in lane, 16 contiguous K per half)
__device__ __forceinline__ v16bf load_afrag(const __bf16* ap) {
  v16bf a;
  ((int4*)&a)[0] = *(const int4*)ap;
  ((int4*)&a)[1] = *(const int4*)(ap + 8);
  return a;
}
// Two ds_load_b128 -> one WMMA B fragment (col in lane, 16 contiguous K per half)
__device__ __forceinline__ v16bf load_bfrag(const __bf16* bp) {
  v16bf b;
  ((int4*)&b)[0] = *(const int4*)bp;
  ((int4*)&b)[1] = *(const int4*)(bp + 8);
  return b;
}

// -------------------------------------------------------------------------
// Kernel 1: Q/K/V projection (1x1 conv == channel matmul) with bf16 WMMA.
//   q -> [b][h][n][d]  (with +pe, *C^-0.25 folded in)
//   k -> [b][h][n][d]
//   v -> [b][h][d][n]
// -------------------------------------------------------------------------
__global__ __launch_bounds__(256) void qkv_proj_kernel(
    const float* __restrict__ x,
    const float* __restrict__ wq, const float* __restrict__ bq,
    const float* __restrict__ wk, const float* __restrict__ bk,
    const float* __restrict__ wv, const float* __restrict__ bv,
    const float* __restrict__ pe,
    __bf16* __restrict__ qs, __bf16* __restrict__ ks, __bf16* __restrict__ vs)
{
  __shared__ __align__(16) __bf16 Ws[CC][CC];       // weight tile, row-major [o][c]
  __shared__ __align__(16) __bf16 Xs[64][CC + 8];   // x tile transposed [token][c]

  const int t    = threadIdx.x;
  const int b    = blockIdx.y;
  const int n0   = blockIdx.x * 64;
  const int w    = t >> 5;        // wave id = output M-tile
  const int l    = t & 31;
  const int lo16 = l & 15;
  const int hi   = l >> 4;

  // Load X tile transposed to bf16: Xs[n][c]
  for (int i = 0; i < 8; ++i) {
    int idx = t + i * 256;                    // 2048 float4 reads
    int c = idx >> 4;
    int g = idx & 15;
    const float4 vx = *(const float4*)(x + ((size_t)b * CC + c) * NN + n0 + g * 4);
    Xs[g * 4 + 0][c] = f2bf(vx.x);
    Xs[g * 4 + 1][c] = f2bf(vx.y);
    Xs[g * 4 + 2][c] = f2bf(vx.z);
    Xs[g * 4 + 3][c] = f2bf(vx.w);
  }

  const float* wmats[3] = {wq, wk, wv};
  const float* bvecs[3] = {bq, bk, bv};

  const int o0 = w * 16 + 8 * hi;   // this lane's 8 contiguous output channels
  const int hh = o0 >> 6;           // head
  const int d0 = o0 & 63;           // head-dim base

  for (int m = 0; m < 3; ++m) {
    __syncthreads();
    const float* wsrc = wmats[m];
    for (int i = 0; i < 16; ++i) {            // 4096 float4 reads
      int idx = t + i * 256;
      int o = idx >> 5;
      int g = idx & 31;
      const float4 vw = *(const float4*)(wsrc + (size_t)o * CC + g * 4);
      uint2 pk;
      pk.x = (unsigned)f2bfu(vw.x) | ((unsigned)f2bfu(vw.y) << 16);
      pk.y = (unsigned)f2bfu(vw.z) | ((unsigned)f2bfu(vw.w) << 16);
      *(uint2*)&Ws[o][g * 4] = pk;
    }
    __syncthreads();

    // Hoist the 4 loop-invariant A (weight) fragments for this wave's M-tile.
    v16bf aw[4];
    #pragma unroll
    for (int kc = 0; kc < 4; ++kc)
      aw[kc] = load_afrag(&Ws[w * 16 + lo16][kc * 32 + hi * 16]);

    const float* bias = bvecs[m];
    for (int nt = 0; nt < 4; ++nt) {
      // Batch all B fragments, then run the WMMA chain.
      v16bf bx[4];
      #pragma unroll
      for (int kc = 0; kc < 4; ++kc)
        bx[kc] = load_bfrag(&Xs[nt * 16 + lo16][kc * 32 + hi * 16]);
      v8f acc = {};
      #pragma unroll
      for (int kc = 0; kc < 4; ++kc)        // K = 128 = 4 x 32
        acc = __builtin_amdgcn_wmma_f32_16x16x32_bf16(false, aw[kc], false, bx[kc],
                                                      (short)0, acc, false, false);

      const int n = n0 + nt * 16 + lo16;
      float vals[8];
      #pragma unroll
      for (int r = 0; r < 8; ++r) vals[r] = acc[r] + bias[o0 + r];
      if (m == 0) {
        #pragma unroll
        for (int r = 0; r < 8; ++r)
          vals[r] = (vals[r] + pe[((size_t)n * DD + d0 + r) * NHD + hh]) * QSCALE;
      }
      if (m < 2) {
        // [b][h][n][d] : 8 contiguous bf16 -> one b128 store
        unsigned short p8[8];
        #pragma unroll
        for (int r = 0; r < 8; ++r) p8[r] = f2bfu(vals[r]);
        int4 st;
        st.x = p8[0] | (p8[1] << 16);
        st.y = p8[2] | (p8[3] << 16);
        st.z = p8[4] | (p8[5] << 16);
        st.w = p8[6] | (p8[7] << 16);
        __bf16* dst = (m == 0) ? qs : ks;
        *(int4*)(dst + (((size_t)b * NHD + hh) * NN + n) * DD + d0) = st;
      } else {
        // V: [b][h][d][n]
        #pragma unroll
        for (int r = 0; r < 8; ++r)
          vs[(((size_t)b * NHD + hh) * DD + d0 + r) * NN + n] = f2bf(vals[r]);
      }
    }
  }
}

// -------------------------------------------------------------------------
// Kernel 2: flash attention. Block = 128 queries x (b,h); wave owns 16 rows.
// -------------------------------------------------------------------------
__global__ __launch_bounds__(256) void attn_kernel(
    const __bf16* __restrict__ qs, const __bf16* __restrict__ ks,
    const __bf16* __restrict__ vs,
    float* __restrict__ out0, float* __restrict__ out1)
{
  __shared__ __align__(16) __bf16 Qs[128][72];  // [query][d]
  __shared__ __align__(16) __bf16 Ks[64][72];   // [key][d]
  __shared__ __align__(16) __bf16 Vs[64][72];   // [d][key]
  __shared__ __align__(16) __bf16 Ps[8][16][72];// per-wave P staging [q][key]

  const int t    = threadIdx.x;
  const int h    = blockIdx.y;
  const int b    = blockIdx.z;
  const int q0   = blockIdx.x * 128;
  const int w    = t >> 5;
  const int l    = t & 31;
  const int lo16 = l & 15;
  const int hi   = l >> 4;

  const __bf16* qbase = qs + ((size_t)b * NHD + h) * NN * DD;
  const __bf16* kbase = ks + ((size_t)b * NHD + h) * NN * DD;
  const __bf16* vbase = vs + ((size_t)b * NHD + h) * DD * NN;

  // Load Q tile (128 x 64 bf16)
  for (int i = 0; i < 4; ++i) {
    int idx = t + i * 256;
    int row = idx >> 3, seg = idx & 7;
    *(int4*)&Qs[row][seg * 8] =
        *(const int4*)(qbase + (size_t)(q0 + row) * DD + seg * 8);
  }
  __syncthreads();

  // Q A-fragments are invariant over the whole key loop: load them ONCE.
  v16bf aQ[2];
  #pragma unroll
  for (int kc = 0; kc < 2; ++kc)
    aQ[kc] = load_afrag(&Qs[w * 16 + lo16][kc * 32 + hi * 16]);

  float mst[8], lst[8];
  v8f accO[4] = {};
  #pragma unroll
  for (int r = 0; r < 8; ++r) { mst[r] = -1e30f; lst[r] = 0.f; }

  const int krow = t >> 3, kseg = t & 7;   // this thread's tile-staging slot

  for (int kt = 0; kt < NN / 64; ++kt) {
    __syncthreads();                       // previous tile fully consumed
    // Stage 64x64 K and V tiles into LDS.
#if USE_ASYNC_LDS
    #pragma unroll
    for (int i = 0; i < 2; ++i) {
      int idx = t + i * 256;
      int row = idx >> 3, seg = idx & 7;
      async_copy16(kbase + (size_t)(kt * 64 + row) * DD + seg * 8,
                   &Ks[row][seg * 8]);
      async_copy16(vbase + (size_t)row * NN + kt * 64 + seg * 8,
                   &Vs[row][seg * 8]);
    }
    wait_async0();
#else
    #pragma unroll
    for (int i = 0; i < 2; ++i) {
      int idx = t + i * 256;
      int row = idx >> 3, seg = idx & 7;
      *(int4*)&Ks[row][seg * 8] =
          *(const int4*)(kbase + (size_t)(kt * 64 + row) * DD + seg * 8);
      *(int4*)&Vs[row][seg * 8] =
          *(const int4*)(vbase + (size_t)row * NN + kt * 64 + seg * 8);
    }
#endif
    __syncthreads();

    if (kt + 1 < NN / 64) {                // prefetch next tiles into caches
      __builtin_prefetch(kbase + (size_t)((kt + 1) * 64 + krow) * DD + kseg * 8, 0, 0);
      __builtin_prefetch(vbase + (size_t)krow * NN + (kt + 1) * 64 + kseg * 8, 0, 0);
    }

    // ---- Scores: wave's 16 queries x 64 keys (d = 64 = 2 x K32) ----
    v16bf kf[8];
    #pragma unroll
    for (int nt = 0; nt < 4; ++nt)
      #pragma unroll
      for (int kc = 0; kc < 2; ++kc)
        kf[nt * 2 + kc] = load_bfrag(&Ks[nt * 16 + lo16][kc * 32 + hi * 16]);

    v8f sc[4];
    #pragma unroll
    for (int nt = 0; nt < 4; ++nt) {
      v8f s = {};
      s = __builtin_amdgcn_wmma_f32_16x16x32_bf16(false, aQ[0], false, kf[nt * 2 + 0],
                                                  (short)0, s, false, false);
      s = __builtin_amdgcn_wmma_f32_16x16x32_bf16(false, aQ[1], false, kf[nt * 2 + 1],
                                                  (short)0, s, false, false);
      sc[nt] = s;
    }

    // Preload V B-fragments now so DS latency overlaps the softmax VALU work.
    v16bf vf[8];
    #pragma unroll
    for (int dt = 0; dt < 4; ++dt)
      #pragma unroll
      for (int kc = 0; kc < 2; ++kc)
        vf[dt * 2 + kc] = load_bfrag(&Vs[dt * 16 + lo16][kc * 32 + hi * 16]);

    // ---- Online softmax. C layout: row m = r + 8*hi, col = lo16 ----
    #pragma unroll
    for (int r = 0; r < 8; ++r) {
      float tm = fmaxf(fmaxf(sc[0][r], sc[1][r]), fmaxf(sc[2][r], sc[3][r]));
      #pragma unroll
      for (int off = 1; off < 16; off <<= 1)
        tm = fmaxf(tm, __shfl_xor(tm, off, 32));
      float mnew  = fmaxf(mst[r], tm);
      float alpha = __expf(mst[r] - mnew);
      float psum  = 0.f;
      #pragma unroll
      for (int nt = 0; nt < 4; ++nt) {
        float p = __expf(sc[nt][r] - mnew);
        sc[nt][r] = p;
        psum += p;
      }
      #pragma unroll
      for (int off = 1; off < 16; off <<= 1)
        psum += __shfl_xor(psum, off, 32);
      lst[r] = lst[r] * alpha + psum;
      mst[r] = mnew;
      #pragma unroll
      for (int dt = 0; dt < 4; ++dt) accO[dt][r] *= alpha;   // rescale O
    }

    // Stage P (C layout) -> LDS row-major [q][key] for A fragments
    #pragma unroll
    for (int nt = 0; nt < 4; ++nt)
      #pragma unroll
      for (int r = 0; r < 8; ++r)
        Ps[w][r + 8 * hi][nt * 16 + lo16] = f2bf(sc[nt][r]);

    v16bf aP[2];
    #pragma unroll
    for (int kc = 0; kc < 2; ++kc)
      aP[kc] = load_afrag(&Ps[w][lo16][kc * 32 + hi * 16]);

    // ---- O update: oT[m][d] += P[m][key] * V[d][key]^T ----
    #pragma unroll
    for (int dt = 0; dt < 4; ++dt) {
      accO[dt] = __builtin_amdgcn_wmma_f32_16x16x32_bf16(false, aP[0], false, vf[dt * 2 + 0],
                                                         (short)0, accO[dt], false, false);
      accO[dt] = __builtin_amdgcn_wmma_f32_16x16x32_bf16(false, aP[1], false, vf[dt * 2 + 1],
                                                         (short)0, accO[dt], false, false);
    }
  }

  // Finalize: out channel c = d*NH + h ; write both identical output copies.
  float inv[8];
  #pragma unroll
  for (int r = 0; r < 8; ++r) inv[r] = (lst[r] > 0.f) ? 1.f / lst[r] : 0.f;

  #pragma unroll
  for (int dt = 0; dt < 4; ++dt) {
    int d  = dt * 16 + lo16;
    int nq = q0 + w * 16 + 8 * hi;          // +r is contiguous tokens
    size_t basei = (((size_t)b * DD + d) * NHD + h) * (size_t)NN + nq;
    float4 lo4, hi4;
    lo4.x = accO[dt][0] * inv[0]; lo4.y = accO[dt][1] * inv[1];
    lo4.z = accO[dt][2] * inv[2]; lo4.w = accO[dt][3] * inv[3];
    hi4.x = accO[dt][4] * inv[4]; hi4.y = accO[dt][5] * inv[5];
    hi4.z = accO[dt][6] * inv[6]; hi4.w = accO[dt][7] * inv[7];
    *(float4*)(out0 + basei)     = lo4;
    *(float4*)(out0 + basei + 4) = hi4;
    *(float4*)(out1 + basei)     = lo4;
    *(float4*)(out1 + basei + 4) = hi4;
  }
}

// -------------------------------------------------------------------------
extern "C" void kernel_launch(void* const* d_in, const int* in_sizes, int n_in,
                              void* d_out, int out_size, void* d_ws, size_t ws_size,
                              hipStream_t stream) {
  const float* x  = (const float*)d_in[0];
  const float* wq = (const float*)d_in[1];
  const float* bq = (const float*)d_in[2];
  const float* wk = (const float*)d_in[3];
  const float* bk = (const float*)d_in[4];
  const float* wv = (const float*)d_in[5];
  const float* bv = (const float*)d_in[6];
  const float* pe = (const float*)d_in[7];

  const size_t elems = (size_t)BB * NHD * NN * DD;   // 2,097,152 bf16 each
  __bf16* qs = (__bf16*)d_ws;
  __bf16* ks = qs + elems;
  __bf16* vs = ks + elems;

  float* out0 = (float*)d_out;                 // out   [B,C,H,W]
  float* out1 = out0 + (size_t)BB * CC * NN;   // attn_out (same flat data)

  dim3 gp(NN / 64, BB);                        // (64, 4) blocks
  qkv_proj_kernel<<<gp, 256, 0, stream>>>(x, wq, bq, wk, bk, wv, bv, pe,
                                          qs, ks, vs);

  dim3 ga(NN / 128, NHD, BB);                  // (32, 2, 4) blocks
  attn_kernel<<<ga, 256, 0, stream>>>(qs, ks, vs, out0, out1);
}
